// CausalSelfAttention_masked_for_formula_88055419502957
// MI455X (gfx1250) — matrix operand order repaired and checked by reading, not compiled, not run on hardware
//
#include <hip/hip_runtime.h>
#include <stdint.h>

// MI455X / gfx1250, wave32. Compile-only target.
// Workspace layout (needs >= 56 MB):
//   [ 0MB) Wt   : 4 transposed fp32 weights (N-major), 4 x 4MB
//   [16MB) Qh   : f16 [B,H,T,D]   8MB
//   [24MB) Kh   : f16 [B,H,T,D]   8MB
//   [32MB) Vtws : f16 [B,H,D,T]   8MB   (transposed for PV B-fragments)
//   [40MB) Yf   : f32 [B*T, C]   16MB   (attention output, pre-projection)

typedef __attribute__((ext_vector_type(2)))  float    v2f;
typedef __attribute__((ext_vector_type(8)))  float    v8f;
typedef __attribute__((ext_vector_type(16))) _Float16 v16h;

static constexpr int Bq = 4, Tq = 1024, Cq = 1024, Hq = 16, Dq = 64;

// ---------------- WMMA wrappers ----------------
__device__ inline v8f wmma_f16(v16h a, v16h b, v8f c) {
  // D = A(16x32 f16) x B(32x16 f16) + C(16x16 f32)
  return __builtin_amdgcn_wmma_f32_16x16x32_f16(false, a, false, b, (short)0, c,
                                                false, false);
}
__device__ inline v8f wmma_f32(v2f a, v2f b, v8f c) {
  // D = A(16x4 f32) x B(4x16 f32) + C(16x16 f32)  -- exact fp32 path
  return __builtin_amdgcn_wmma_f32_16x16x4_f32(false, a, false, b, (short)0, c,
                                               false, false);
}

// ---------------- CDNA5 async global->LDS copy (ASYNCcnt path) --------------
// Generic LDS pointers carry the LDS byte offset in their low 32 bits
// (flat->LDS mapping truncates to addr[31:0], ISA 10.2).
__device__ inline uint32_t lds_off(const void* p) {
  return (uint32_t)(uintptr_t)p;
}
__device__ inline void async_g2l_b128(uint32_t ldsaddr, const void* gaddr) {
  // per-lane: LDS[ldsaddr] = MEM[gaddr], 16 bytes; tracked by ASYNCcnt
  asm volatile("global_load_async_to_lds_b128 %0, %1, off"
               :: "v"(ldsaddr), "v"(gaddr) : "memory");
}
__device__ inline void wait_async0() {
  asm volatile("s_wait_asynccnt 0" ::: "memory");
}

// ---------------- f16 fragment loaders (wave32, contiguous k order, same for A and B) ----------
// A 16x32 (MxK), row-major source with leading dim ld:
//   lane&15 = row M; (lane>>4)*16 = K base; elems e -> K kb+e (contiguous).
__device__ inline v16h ld_a16(const _Float16* base, int ld, int lane) {
  const int r  = lane & 15;
  const int kb = (lane >> 4) << 4;
  const _Float16* p = base + r * ld + kb;
  v16h f;
#pragma unroll
  for (int i = 0; i < 16; ++i) f[i] = p[i];
  return f;
}
// B 32x16 (KxN), source is N-major (bT[n*ld + k]):
//   lane&15 = col N; (lane>>4)*16 = K base; elems e -> K kb+e (contiguous).
__device__ inline v16h ld_b16(const _Float16* bT, int ld, int lane) {
  const int n  = lane & 15;
  const int kb = (lane >> 4) << 4;
  const _Float16* p = bT + n * ld + kb;
  v16h f;
#pragma unroll
  for (int i = 0; i < 16; ++i) f[i] = p[i];
  return f;
}

// 16-lane reductions (rows 0..7 live in lanes 0-15, rows 8..15 in lanes 16-31)
__device__ inline float redmax16(float v) {
#pragma unroll
  for (int off = 1; off < 16; off <<= 1) v = fmaxf(v, __shfl_xor(v, off, 32));
  return v;
}
__device__ inline float redsum16(float v) {
#pragma unroll
  for (int off = 1; off < 16; off <<= 1) v += __shfl_xor(v, off, 32);
  return v;
}

// ---------------- weight transpose: Wt[n*C + k] = W[k*C + n] ----------------
__global__ __launch_bounds__(256)
void transpose_w(const float* __restrict__ Wqp, const float* __restrict__ Wkp,
                 const float* __restrict__ Wvp, const float* __restrict__ Wpp,
                 float* __restrict__ Wt) {
  const int w = blockIdx.y;
  const float* src = (w == 0) ? Wqp : (w == 1) ? Wkp : (w == 2) ? Wvp : Wpp;
  const size_t idx = (size_t)blockIdx.x * 256 + threadIdx.x;  // over C*C
  const int nn = (int)(idx >> 10);
  const int kk = (int)(idx & 1023);
  Wt[(size_t)w * Cq * Cq + idx] = src[(size_t)kk * Cq + nn];  // coalesced write
}

// ---------------- fp32 WMMA GEMM: Y[M,N] = X[M,K] * Wt^T + bias -------------
// MODE 0: Q -> f16 [B,H,T,D]; MODE 1: K -> f16 [B,H,T,D];
// MODE 2: V -> f16 [B,H,D,T]; MODE 3: out -> f32 row-major [M,N].
// Block = 4 waves; block tile 64(M) x 64(N); K staged in 64-wide chunks in
// LDS via CDNA5 async global->LDS copies; WMMA fragments fed by ds_load.
template <int MODE>
__global__ __launch_bounds__(128)
void proj_gemm(const float* __restrict__ X, const float* __restrict__ Wt,
               const float* __restrict__ bias, void* __restrict__ outv) {
  __shared__ float Xs[64 * 64];  // 16 KB  [m][k] chunk, row-major, ld=64
  __shared__ float Ws[64 * 64];  // 16 KB  [n][k] chunk, N-major,  ld=64

  const int tid  = threadIdx.x;
  const int lane = tid & 31;
  const int wave = tid >> 5;
  const int m0   = blockIdx.x * 64;
  const int n0   = blockIdx.y * 64;
  const int r16  = lane & 15;
  const int kb   = (lane >> 4) << 1;  // K base 0 or 2 (A and B, K=4)

  const float* pa  = Xs + (wave * 16 + r16) * 64 + kb;
  const float* pb0 = Ws + ( 0 + r16) * 64 + kb;
  const float* pb1 = Ws + (16 + r16) * 64 + kb;
  const float* pb2 = Ws + (32 + r16) * 64 + kb;
  const float* pb3 = Ws + (48 + r16) * 64 + kb;

  v8f acc0 = {}, acc1 = {}, acc2 = {}, acc3 = {};

  for (int kc = 0; kc < Cq; kc += 64) {
    // stage 64x64 X and W chunks: 1024 x 16B segments each, 8 per thread,
    // consecutive tids hit consecutive 16B segments (coalesced).
#pragma unroll
    for (int i = 0; i < 8; ++i) {
      const int s   = i * 128 + tid;
      const int row = s >> 4;        // 16 segments per 64-float row
      const int seg = (s & 15) << 2; // float offset within row
      async_g2l_b128(lds_off(&Xs[row * 64 + seg]),
                     X + (size_t)(m0 + row) * Cq + kc + seg);
      async_g2l_b128(lds_off(&Ws[row * 64 + seg]),
                     Wt + (size_t)(n0 + row) * Cq + kc + seg);
    }
    wait_async0();      // our copies have landed in LDS
    __syncthreads();    // everyone's copies visible

#pragma unroll
    for (int k = 0; k < 64; k += 4) {
      v2f a  = {pa[k],  pa[k + 1]};
      v2f b0 = {pb0[k], pb0[k + 1]};
      v2f b1 = {pb1[k], pb1[k + 1]};
      v2f b2 = {pb2[k], pb2[k + 1]};
      v2f b3 = {pb3[k], pb3[k + 1]};
      acc0 = wmma_f32(a, b0, acc0);
      acc1 = wmma_f32(a, b1, acc1);
      acc2 = wmma_f32(a, b2, acc2);
      acc3 = wmma_f32(a, b3, acc3);
    }
    __syncthreads();    // done reading before next chunk overwrites
  }

  v8f accs[4] = {acc0, acc1, acc2, acc3};
  const int mhi = (lane >> 4) << 3;
#pragma unroll
  for (int j = 0; j < 4; ++j) {
    const int nn = n0 + j * 16 + r16;
    const float bv = bias[nn];
#pragma unroll
    for (int r = 0; r < 8; ++r) {
      const int mm = m0 + wave * 16 + mhi + r;
      const float val = accs[j][r] + bv;
      if (MODE == 3) {
        ((float*)outv)[(size_t)mm * Cq + nn] = val;
      } else {
        const int bb = mm >> 10, tt = mm & 1023;
        const int hh = nn >> 6,  dd = nn & 63;
        _Float16* o = (_Float16*)outv;
        if (MODE == 2)
          o[(((size_t)(bb * Hq + hh)) * Dq + dd) * Tq + tt] = (_Float16)val;
        else
          o[(((size_t)(bb * Hq + hh)) * Tq + tt) * Dq + dd] = (_Float16)val;
      }
    }
  }
}

// ---------------- flash attention (padding mask, no causal) -----------------
// grid = (T/64 q-tiles, B*H). Block = 4 waves; each wave owns 16 q rows.
__global__ __launch_bounds__(128)
void attn_flash(const _Float16* __restrict__ Qh, const _Float16* __restrict__ Kh,
                const _Float16* __restrict__ Vt, const float* __restrict__ pad,
                float* __restrict__ Yf) {
  __shared__ float keep[Tq];
  __shared__ _Float16 Pl[4][16 * 32];  // per-wave P tile (C-layout -> A-layout)

  const int bh = blockIdx.y;
  const int b  = bh >> 4;
  const int h  = bh & 15;
  const int q0 = blockIdx.x * 64;

  for (int i = threadIdx.x; i < Tq; i += 128)
    keep[i] = 1.0f - pad[b * Tq + i];
  __syncthreads();

  const int lane = threadIdx.x & 31;
  const int wave = threadIdx.x >> 5;
  const int qm   = q0 + wave * 16;

  const _Float16* Qp = Qh + (size_t)bh * Tq * Dq;
  const _Float16* Kp = Kh + (size_t)bh * Tq * Dq;
  const _Float16* Vp = Vt + (size_t)bh * Dq * Tq;

  const v16h qa0 = ld_a16(Qp + qm * Dq + 0,  Dq, lane);
  const v16h qa1 = ld_a16(Qp + qm * Dq + 32, Dq, lane);

  const int n   = lane & 15;
  const int mhi = (lane >> 4) << 3;

  float kq[8];
#pragma unroll
  for (int r = 0; r < 8; ++r) kq[r] = keep[qm + mhi + r];

  float rmax[8], rsum[8];
#pragma unroll
  for (int r = 0; r < 8; ++r) { rmax[r] = -3.0e38f; rsum[r] = 0.0f; }
  v8f o0 = {}, o1 = {}, o2 = {}, o3 = {};

  _Float16* P = &Pl[wave][0];

  for (int kt = 0; kt < Tq; kt += 32) {
    // prefetch next tiles (L2-resident; WGP-scope, -> global_prefetch_b8)
    if (kt + 32 < Tq) {
      __builtin_prefetch(Kp + (size_t)(kt + 32) * Dq + lane * 64, 0, 0);
      __builtin_prefetch(Vp + (size_t)(lane * 2) * Tq + kt + 32, 0, 0);
    }

    // S(16x32) = Q(16x64) * K^T; K rows are d-contiguous -> already N-major B
    v8f s0 = {}, s1 = {};
    s0 = wmma_f16(qa0, ld_b16(Kp + (size_t)(kt +  0) * Dq + 0,  Dq, lane), s0);
    s0 = wmma_f16(qa1, ld_b16(Kp + (size_t)(kt +  0) * Dq + 32, Dq, lane), s0);
    s1 = wmma_f16(qa0, ld_b16(Kp + (size_t)(kt + 16) * Dq + 0,  Dq, lane), s1);
    s1 = wmma_f16(qa1, ld_b16(Kp + (size_t)(kt + 16) * Dq + 32, Dq, lane), s1);

    const float kk0 = keep[kt + n];
    const float kk1 = keep[kt + 16 + n];

    float e0[8], e1[8];
#pragma unroll
    for (int r = 0; r < 8; ++r) {
      float v0 = s0[r] * 0.125f;  // 1/sqrt(64)
      float v1 = s1[r] * 0.125f;
      if (kq[r] * kk0 == 0.0f) v0 = -1e9f;  // matches reference masked_fill
      if (kq[r] * kk1 == 0.0f) v1 = -1e9f;
      const float tm = redmax16(fmaxf(v0, v1));
      const float nm = fmaxf(rmax[r], tm);
      const float corr = __expf(rmax[r] - nm);
      rmax[r] = nm;
      const float a0 = __expf(v0 - nm);
      const float a1 = __expf(v1 - nm);
      rsum[r] = rsum[r] * corr + redsum16(a0 + a1);
      o0[r] *= corr; o1[r] *= corr; o2[r] *= corr; o3[r] *= corr;
      e0[r] = a0; e1[r] = a1;
    }

    // transpose P (C-layout regs) -> A-layout via per-wave LDS tile
#pragma unroll
    for (int r = 0; r < 8; ++r) {
      P[(mhi + r) * 32 + n]      = (_Float16)e0[r];
      P[(mhi + r) * 32 + n + 16] = (_Float16)e1[r];
    }
    __builtin_amdgcn_wave_barrier();  // DS ops are in-order within a wave
    const v16h paf = ld_a16(P, 32, lane);
    __builtin_amdgcn_wave_barrier();

    // O(16x64) += P(16x32) * V(32x64); Vt is [d][t] -> N-major B over tokens
    o0 = wmma_f16(paf, ld_b16(Vp + (size_t) 0 * Tq + kt, Tq, lane), o0);
    o1 = wmma_f16(paf, ld_b16(Vp + (size_t)16 * Tq + kt, Tq, lane), o1);
    o2 = wmma_f16(paf, ld_b16(Vp + (size_t)32 * Tq + kt, Tq, lane), o2);
    o3 = wmma_f16(paf, ld_b16(Vp + (size_t)48 * Tq + kt, Tq, lane), o3);
  }

  v8f oo[4] = {o0, o1, o2, o3};
#pragma unroll
  for (int j = 0; j < 4; ++j) {
#pragma unroll
    for (int r = 0; r < 8; ++r) {
      const int t = qm + mhi + r;
      const int c = h * Dq + j * 16 + n;
      Yf[((size_t)(b * Tq + t)) * Cq + c] = oo[j][r] / rsum[r];
    }
  }
}

// ---------------- host entry ----------------
extern "C" void kernel_launch(void* const* d_in, const int* in_sizes, int n_in,
                              void* d_out, int out_size, void* d_ws,
                              size_t ws_size, hipStream_t stream) {
  (void)in_sizes; (void)n_in; (void)out_size; (void)ws_size;
  const float* x   = (const float*)d_in[0];
  const float* pad = (const float*)d_in[1];
  const float* Wq  = (const float*)d_in[2];
  const float* bq  = (const float*)d_in[3];
  const float* Wk  = (const float*)d_in[4];
  const float* bk  = (const float*)d_in[5];
  const float* Wv  = (const float*)d_in[6];
  const float* bv  = (const float*)d_in[7];
  const float* Wp  = (const float*)d_in[8];
  const float* bp  = (const float*)d_in[9];
  float* out = (float*)d_out;

  char* ws = (char*)d_ws;
  float*    Wt = (float*)ws;                      // 16 MB
  _Float16* Qh = (_Float16*)(ws + (16u << 20));   //  8 MB
  _Float16* Kh = (_Float16*)(ws + (24u << 20));   //  8 MB
  _Float16* Vw = (_Float16*)(ws + (32u << 20));   //  8 MB
  float*    Yf = (float*)   (ws + (40u << 20));   // 16 MB

  transpose_w<<<dim3(Cq * Cq / 256, 4), 256, 0, stream>>>(Wq, Wk, Wv, Wp, Wt);

  dim3 g(Bq * Tq / 64, Cq / 64);
  proj_gemm<0><<<g, 128, 0, stream>>>(x, Wt + (size_t)0 * Cq * Cq, bq, Qh);
  proj_gemm<1><<<g, 128, 0, stream>>>(x, Wt + (size_t)1 * Cq * Cq, bk, Kh);
  proj_gemm<2><<<g, 128, 0, stream>>>(x, Wt + (size_t)2 * Cq * Cq, bv, Vw);

  attn_flash<<<dim3(Tq / 64, Bq * Hq), 128, 0, stream>>>(Qh, Kh, Vw, pad, Yf);

  proj_gemm<3><<<g, 128, 0, stream>>>(Yf, Wt + (size_t)3 * Cq * Cq, bp, out);
}
